// GradientLayerPoisson_46213848105284
// MI455X (gfx1250) — matrix operand order repaired and checked by reading, not compiled, not run on hardware
//
#include <hip/hip_runtime.h>

// PINN tanh-MLP with forward-over-forward JVP (u, du/dx0, d2u/dx0^2).
// Compute-bound (~103 GFLOP vs ~4 MB HBM traffic) -> WMMA f16 x f16 -> f32.

typedef _Float16 half8  __attribute__((ext_vector_type(8)));
typedef _Float16 half16 __attribute__((ext_vector_type(16)));
typedef float    float8 __attribute__((ext_vector_type(8)));

#define HID     256
#define BATCH   131072
#define MT      16                  // samples per M-tile (WMMA M)
#define NWAVES  16                  // one wave per 16-column N-tile
#define THREADS (NWAVES * 32)
#define KSTEPS  (HID / 32)          // 8 WMMA K-steps of 32
#define AROW    (HID + 8)           // pad LDS rows by 16B to rotate banks

__global__ __launch_bounds__(THREADS, 1)
void GradientLayerPoisson_kernel(const float* __restrict__ x,
                                 const float* __restrict__ W1, const float* __restrict__ b1,
                                 const float* __restrict__ W2, const float* __restrict__ b2,
                                 const float* __restrict__ W3, const float* __restrict__ b3,
                                 const float* __restrict__ W4, const float* __restrict__ b4,
                                 float* __restrict__ out)
{
    // activations for (h, dh, d2h): 3 * 16 * 264 * 2B = ~25 KB LDS
    __shared__ __align__(16) _Float16 sAct[3][MT][AROW];

    const int tid   = threadIdx.x;
    const int wave  = tid >> 5;
    const int lane  = tid & 31;
    const int lrow  = lane & 15;                  // row (A) / col-in-tile (B,C)
    const int akoff = (lane < 16) ? 0 : 16;       // 16-bit A-matrix K sub-offset
    const int bkoff = (lane < 16) ? 0 : 16;       // 16-bit B-matrix K sub-offset
    const int col   = wave * 16 + lrow;           // this lane's hidden-unit column

    // ---- Stage W2/W3 B-fragments in registers (f32 -> f16), once per WG ----
    // B (32x16 f16): lane l holds column (l&15), 16 contiguous K values
    // starting at (l<16 ? 0 : 16) within the 32-wide K step.
    half16 w2f[KSTEPS], w3f[KSTEPS];
#pragma unroll
    for (int ks = 0; ks < KSTEPS; ++ks) {
        half16 f2, f3;
#pragma unroll
        for (int i = 0; i < 16; ++i) {
            const int k = ks * 32 + bkoff + i;
            f2[i] = (_Float16)W2[k * HID + col];
            f3[i] = (_Float16)W3[k * HID + col];
        }
        w2f[ks] = f2;
        w3f[ks] = f3;
    }
    const float bias2 = b2[col];
    const float bias3 = b3[col];
    const float b4s   = b4[0];
    float w4r[8];
#pragma unroll
    for (int t = 0; t < 8; ++t) w4r[t] = W4[lane + 32 * t];

    const int nTiles = BATCH / MT;                // 8192
    for (int tile = blockIdx.x; tile < nTiles; tile += gridDim.x) {
        const int m0 = tile * MT;

        // ---- Layer 1: (x @ W1 + b1), tanh + first/second tangents ----
        for (int idx = tid; idx < MT * HID; idx += THREADS) {
            const int s = idx >> 8;
            const int j = idx & 255;
            const float x0 = x[(m0 + s) * 2 + 0];
            const float x1 = x[(m0 + s) * 2 + 1];
            const float w0 = W1[j];               // row 0 => dz1/dx0
            const float w1 = W1[HID + j];
            const float z  = fmaf(x0, w0, fmaf(x1, w1, b1[j]));
            const float h  = tanhf(z);
            const float tp = 1.0f - h * h;
            const float dz = w0;
            const float dh = tp * dz;
            const float d2 = -2.0f * h * tp * dz * dz;   // d2z1 == 0
            sAct[0][s][j] = (_Float16)h;
            sAct[1][s][j] = (_Float16)dh;
            sAct[2][s][j] = (_Float16)d2;
        }
        __syncthreads();

        // ---- Layers 2 and 3: three WMMA streams sharing B-fragments ----
#pragma unroll
        for (int layer = 0; layer < 2; ++layer) {
            float8 accH = {}, accD = {}, accS = {};
#pragma unroll
            for (int ks = 0; ks < KSTEPS; ++ks) {
                const half16 bf = (layer == 0) ? w2f[ks] : w3f[ks];
                const int kb = ks * 32 + akoff;
                // A (16x32 f16): lane row = lrow; halves [0..15] = K kb..kb+15
                // (two 16B LDS loads each).
                const _Float16* pH = &sAct[0][lrow][kb];
                const _Float16* pD = &sAct[1][lrow][kb];
                const _Float16* pS = &sAct[2][lrow][kb];
                half8 l0 = *(const half8*)(pH);
                half8 h0 = *(const half8*)(pH + 8);
                half8 l1 = *(const half8*)(pD);
                half8 h1 = *(const half8*)(pD + 8);
                half8 l2 = *(const half8*)(pS);
                half8 h2 = *(const half8*)(pS + 8);
                half16 aH = __builtin_shufflevector(l0, h0, 0,1,2,3,4,5,6,7,8,9,10,11,12,13,14,15);
                half16 aD = __builtin_shufflevector(l1, h1, 0,1,2,3,4,5,6,7,8,9,10,11,12,13,14,15);
                half16 aS = __builtin_shufflevector(l2, h2, 0,1,2,3,4,5,6,7,8,9,10,11,12,13,14,15);
                accH = __builtin_amdgcn_wmma_f32_16x16x32_f16(false, aH, false, bf,
                                                              (short)0, accH, false, false);
                accD = __builtin_amdgcn_wmma_f32_16x16x32_f16(false, aD, false, bf,
                                                              (short)0, accD, false, false);
                accS = __builtin_amdgcn_wmma_f32_16x16x32_f16(false, aS, false, bf,
                                                              (short)0, accS, false, false);
            }
            __syncthreads();   // all waves done READING layer-l activations

            // C/D layout: VGPR r -> M = r (+8 for lanes 16-31), N = col
            const float bz = (layer == 0) ? bias2 : bias3;
            const int mbase = (lane < 16) ? 0 : 8;
#pragma unroll
            for (int r = 0; r < 8; ++r) {
                const float z   = accH[r] + bz;
                const float h   = tanhf(z);
                const float tp  = 1.0f - h * h;
                const float dz  = accD[r];
                const float d2z = accS[r];
                const float dh  = tp * dz;
                const float d2h = tp * d2z - 2.0f * h * tp * dz * dz;
                sAct[0][mbase + r][col] = (_Float16)h;
                sAct[1][mbase + r][col] = (_Float16)dh;
                sAct[2][mbase + r][col] = (_Float16)d2h;
            }
            __syncthreads();
        }

        // ---- Layer 4: wave w reduces sample w against W4 (256 -> 1) ----
        {
            const int s = wave;
            float su = 0.f, sd = 0.f, s2 = 0.f;
#pragma unroll
            for (int t = 0; t < 8; ++t) {
                const int j = lane + 32 * t;
                const float w4 = w4r[t];
                su += (float)sAct[0][s][j] * w4;
                sd += (float)sAct[1][s][j] * w4;
                s2 += (float)sAct[2][s][j] * w4;
            }
#pragma unroll
            for (int off = 16; off > 0; off >>= 1) {
                su += __shfl_xor(su, off, 32);
                sd += __shfl_xor(sd, off, 32);
                s2 += __shfl_xor(s2, off, 32);
            }
            if (lane == 0) {
                out[m0 + s]             = su + b4s;   // u
                out[BATCH + m0 + s]     = sd;         // du/dx0
                out[2 * BATCH + m0 + s] = s2;         // d2u/dx0^2
            }
        }
        __syncthreads();   // before next tile overwrites sAct
    }
}

extern "C" void kernel_launch(void* const* d_in, const int* in_sizes, int n_in,
                              void* d_out, int out_size, void* d_ws, size_t ws_size,
                              hipStream_t stream) {
    const float* x  = (const float*)d_in[0];
    const float* W1 = (const float*)d_in[1];
    const float* b1 = (const float*)d_in[2];
    const float* W2 = (const float*)d_in[3];
    const float* b2 = (const float*)d_in[4];
    const float* W3 = (const float*)d_in[5];
    const float* b3 = (const float*)d_in[6];
    const float* W4 = (const float*)d_in[7];
    const float* b4 = (const float*)d_in[8];
    float* out = (float*)d_out;

    dim3 grid(2048);        // 8192 M-tiles, 4 per workgroup
    dim3 block(THREADS);
    hipLaunchKernelGGL(GradientLayerPoisson_kernel, grid, block, 0, stream,
                       x, W1, b1, W2, b2, W3, b3, W4, b4, out);
}
